// SelfAttention1D_88184268522326
// MI455X (gfx1250) — compile-verified, matrix operand order checked by reading, not run on hardware
//
#include <hip/hip_runtime.h>

// Problem dims (fixed by the reference)
#define DIN   512
#define DOUT  512
#define TSEQ  2048
#define BATCH 8
#define NROWS (BATCH * TSEQ)   // 16384

typedef __attribute__((ext_vector_type(16))) __bf16 v16bf;
typedef __attribute__((ext_vector_type(8)))  __bf16 v8bf;
typedef __attribute__((ext_vector_type(2)))  __bf16 v2bf;
typedef __attribute__((ext_vector_type(8)))  float  v8f;
typedef __attribute__((ext_vector_type(4)))  float  v4f;

// x^p by squaring (p is uniform, =8 at runtime)
__device__ __forceinline__ float ipow_f(float x, int p) {
  float r = 1.0f, b = x;
  unsigned e = (unsigned)p;
  while (e) { if (e & 1u) r *= b; b *= b; e >>= 1u; }
  return r;
}

__device__ __forceinline__ float dot8(v4f q0, v4f q1, const float* __restrict__ krow) {
  v4f k0 = *(const v4f*)(krow);
  v4f k1 = *(const v4f*)(krow + 4);
  float s = q0[0] * k0[0];
  s = fmaf(q0[1], k0[1], s);
  s = fmaf(q0[2], k0[2], s);
  s = fmaf(q0[3], k0[3], s);
  s = fmaf(q1[0], k1[0], s);
  s = fmaf(q1[1], k1[1], s);
  s = fmaf(q1[2], k1[2], s);
  s = fmaf(q1[3], k1[3], s);
  return s;
}

// 16 f32 -> v16bf in A/B fragment element order
__device__ __forceinline__ v16bf cvt16bf(v4f f0, v4f f1, v4f f2, v4f f3) {
  v16bf a;
  a[0]  = (__bf16)f0[0]; a[1]  = (__bf16)f0[1]; a[2]  = (__bf16)f0[2]; a[3]  = (__bf16)f0[3];
  a[4]  = (__bf16)f1[0]; a[5]  = (__bf16)f1[1]; a[6]  = (__bf16)f1[2]; a[7]  = (__bf16)f1[3];
  a[8]  = (__bf16)f2[0]; a[9]  = (__bf16)f2[1]; a[10] = (__bf16)f2[2]; a[11] = (__bf16)f2[3];
  a[12] = (__bf16)f3[0]; a[13] = (__bf16)f3[1]; a[14] = (__bf16)f3[2]; a[15] = (__bf16)f3[3];
  return a;
}

// ---------------- Kernel 1: W_v [k][n] f32 -> Wvt [n][k] bf16 ----------------
__global__ __launch_bounds__(256) void prep_wvt(const float* __restrict__ Wv,
                                                __bf16* __restrict__ Wvt) {
  const int idx = blockIdx.x * 256 + threadIdx.x;  // 0..262143
  const int n = idx >> 9;
  const int k = idx & 511;
  Wvt[idx] = (__bf16)Wv[(size_t)k * DOUT + n];
}

// ---------------- Kernel 2: Q,K fp32 projections (D_ATTN = 8) ----------------
__global__ __launch_bounds__(256) void qk_kernel(const float* __restrict__ In,
                                                 const float* __restrict__ Wq,
                                                 const float* __restrict__ Wk,
                                                 float* __restrict__ Q,
                                                 float* __restrict__ K) {
  const int tid = blockIdx.x * 256 + threadIdx.x;  // 0..131071
  const int row = tid >> 3;
  const int f = tid & 7;
  const float* ir = In + (size_t)row * DIN;
  float q = 0.f, k = 0.f;
  for (int d = 0; d < DIN; ++d) {
    const float iv = ir[d];
    q = fmaf(iv, Wq[d * 8 + f], q);
    k = fmaf(iv, Wk[d * 8 + f], k);
  }
  Q[(size_t)row * 8 + f] = q;
  K[(size_t)row * 8 + f] = k;
}

// ------- Kernel 3: V = Input x W_v (bf16 WMMA), stored transposed Vt[b][n][t] -------
// grid (128, 8), block 256 (8 waves). Wave -> one 16-row M tile x 64 N cols.
__global__ __launch_bounds__(256) void gemm_v(const float* __restrict__ In,
                                              const __bf16* __restrict__ Wvt,
                                              __bf16* __restrict__ Vt) {
  const int lane  = threadIdx.x & 31;
  const int wave  = threadIdx.x >> 5;
  const int mtile = blockIdx.x * 8 + wave;   // 0..1023
  const int nbase = blockIdx.y * 64;         // 0..448
  const int mrow  = lane & 15;
  const int khalf = lane >> 4;

  const float* arow = In + (size_t)(mtile * 16 + mrow) * DIN;
  v8f acc[4];
  v8f zero = {};
#pragma unroll
  for (int j = 0; j < 4; ++j) acc[j] = zero;

  for (int kt = 0; kt < DIN; kt += 32) {
    // A fragment: 16x32 bf16, lane m = lane&15, khalf selects K sub-halves
    const float* a0p = arow + kt + khalf * 8;
    v4f f0 = *(const v4f*)(a0p);
    v4f f1 = *(const v4f*)(a0p + 4);
    v4f f2 = *(const v4f*)(a0p + 16);
    v4f f3 = *(const v4f*)(a0p + 20);
    v16bf a = cvt16bf(f0, f1, f2, f3);
#pragma unroll
    for (int j = 0; j < 4; ++j) {
      const __bf16* brow = Wvt + (size_t)(nbase + j * 16 + mrow) * DIN + kt + khalf * 8;
      v8bf b0 = *(const v8bf*)(brow);
      v8bf b1 = *(const v8bf*)(brow + 16);
      v16bf bb = __builtin_shufflevector(b0, b1, 0, 1, 2, 3, 4, 5, 6, 7,
                                         8, 9, 10, 11, 12, 13, 14, 15);
      acc[j] = __builtin_amdgcn_wmma_f32_16x16x32_bf16(
          false, a, false, bb, (short)0, acc[j], false, false);
    }
  }

  // Store D transposed: element (m = r + 8*khalf, n = nbase+16j+(lane&15))
  const int tbase = (mtile * 16) & (TSEQ - 1);
  const int bidx  = (mtile * 16) >> 11;
#pragma unroll
  for (int j = 0; j < 4; ++j) {
    const int n = nbase + j * 16 + mrow;
    __bf16* col = Vt + (size_t)(bidx * DOUT + n) * TSEQ + tbase + khalf * 8;
#pragma unroll
    for (int r = 0; r < 8; ++r) col[r] = (__bf16)acc[j][r];
  }
}

// ---------------- Kernel 4: per-row softmax stats (max, 1/sum) ----------------
// grid 2048, block 256; one wave per row, lane strides over 64 s-positions.
__global__ __launch_bounds__(256) void stats_kernel(const float* __restrict__ Q,
                                                    const float* __restrict__ Kq,
                                                    const int* __restrict__ Pw,
                                                    float* __restrict__ Mrow,
                                                    float* __restrict__ Linv) {
  const int lane = threadIdx.x & 31;
  const int wave = threadIdx.x >> 5;
  const int row  = blockIdx.x * 8 + wave;
  const int bidx = row >> 11;
  const int p = *Pw;

  const float* qp = Q + (size_t)row * 8;
  v4f q0 = *(const v4f*)(qp);
  v4f q1 = *(const v4f*)(qp + 4);
  const float* Kb = Kq + (size_t)bidx * TSEQ * 8;

  float x[64];
  float mmax = -3.4028235e38f;
#pragma unroll
  for (int j = 0; j < 64; ++j) {
    const int s = j * 32 + lane;
    float sc = dot8(q0, q1, Kb + (size_t)s * 8);
    x[j] = ipow_f(sc, p);
    mmax = fmaxf(mmax, x[j]);
  }
#pragma unroll
  for (int off = 16; off > 0; off >>= 1)
    mmax = fmaxf(mmax, __shfl_xor(mmax, off, 32));
  float lsum = 0.f;
#pragma unroll
  for (int j = 0; j < 64; ++j) lsum += __expf(x[j] - mmax);
#pragma unroll
  for (int off = 16; off > 0; off >>= 1)
    lsum += __shfl_xor(lsum, off, 32);
  if (lane == 0) {
    Mrow[row] = mmax;
    Linv[row] = 1.0f / lsum;
  }
}

// ---------------- Kernel 5: O = softmax(P) x V via bf16 WMMA ----------------
// grid 1024, block 256 (8 waves). Block owns 16 rows x 512 cols; wave -> 64 cols.
// P chunk (16x32) computed cooperatively ONCE per block into LDS (double-buffered),
// then each wave reads its WMMA A-fragment with two ds_load_b128.
__global__ __launch_bounds__(256) void pv_kernel(const float* __restrict__ Q,
                                                 const float* __restrict__ Kq,
                                                 const __bf16* __restrict__ Vt,
                                                 const float* __restrict__ Mrow,
                                                 const float* __restrict__ Linv,
                                                 const int* __restrict__ Pw,
                                                 float* __restrict__ Out) {
  __shared__ __bf16 Plds[2][16 * 32];   // 2 x 1KB double-buffered P chunk

  const int tid   = threadIdx.x;
  const int lane  = tid & 31;
  const int wave  = tid >> 5;
  const int tile  = blockIdx.x;      // 0..1023 (16 rows each)
  const int nbase = wave * 64;
  const int mrow  = lane & 15;
  const int khalf = lane >> 4;
  const int bidx  = (tile * 16) >> 11;
  const int p = *Pw;

  // Producer role: this thread computes P[pm][ps..ps+1] of each 16x32 chunk.
  const int pm = tid >> 4;          // 0..15
  const int ps = (tid & 15) * 2;    // 0,2,..,30
  const int prow = tile * 16 + pm;
  const float* pq = Q + (size_t)prow * 8;
  v4f pq0 = *(const v4f*)(pq);
  v4f pq1 = *(const v4f*)(pq + 4);
  const float pmval = Mrow[prow];

  const float*  Kb = Kq + (size_t)bidx * TSEQ * 8;
  const __bf16* Vb = Vt + (size_t)bidx * DOUT * TSEQ;

  v8f acc[4];
  v8f zero = {};
#pragma unroll
  for (int j = 0; j < 4; ++j) acc[j] = zero;

  for (int S = 0; S < TSEQ; S += 32) {
    const int buf = (S >> 5) & 1;
    // ---- produce: 512 P values, 2 per thread, one packed b32 LDS store ----
    {
      const int s0 = S + ps;
      float sc0 = dot8(pq0, pq1, Kb + (size_t)s0 * 8);
      float sc1 = dot8(pq0, pq1, Kb + (size_t)(s0 + 1) * 8);
      v2bf pp;
      pp[0] = (__bf16)__expf(ipow_f(sc0, p) - pmval);
      pp[1] = (__bf16)__expf(ipow_f(sc1, p) - pmval);
      *(v2bf*)&Plds[buf][pm * 32 + ps] = pp;
    }
    __syncthreads();
    // ---- consume: A-fragment from LDS (two ds_load_b128 per lane) ----
    const __bf16* pr = &Plds[buf][mrow * 32 + khalf * 8];
    v8bf a0 = *(const v8bf*)(pr);
    v8bf a1 = *(const v8bf*)(pr + 16);
    v16bf a = __builtin_shufflevector(a0, a1, 0, 1, 2, 3, 4, 5, 6, 7,
                                      8, 9, 10, 11, 12, 13, 14, 15);
#pragma unroll
    for (int j = 0; j < 4; ++j) {
      const __bf16* brow = Vb + (size_t)(nbase + j * 16 + mrow) * TSEQ + S + khalf * 8;
      v8bf b0 = *(const v8bf*)(brow);
      v8bf b1 = *(const v8bf*)(brow + 16);
      v16bf bb = __builtin_shufflevector(b0, b1, 0, 1, 2, 3, 4, 5, 6, 7,
                                         8, 9, 10, 11, 12, 13, 14, 15);
      acc[j] = __builtin_amdgcn_wmma_f32_16x16x32_bf16(
          false, a, false, bb, (short)0, acc[j], false, false);
    }
    // No second barrier needed: next iteration writes the OTHER buffer; the
    // write that reuses this buffer (two chunks later) is ordered after this
    // chunk's reads by the next iteration's __syncthreads().
  }

  // Epilogue: scale by 1/l per row and store f32 output [row, n]
  float li[8];
#pragma unroll
  for (int r = 0; r < 8; ++r) li[r] = Linv[tile * 16 + r + 8 * khalf];
#pragma unroll
  for (int j = 0; j < 4; ++j) {
    const int n = nbase + j * 16 + mrow;
#pragma unroll
    for (int r = 0; r < 8; ++r) {
      const int rg = tile * 16 + r + 8 * khalf;
      Out[(size_t)rg * DOUT + n] = acc[j][r] * li[r];
    }
  }
}

extern "C" void kernel_launch(void* const* d_in, const int* in_sizes, int n_in,
                              void* d_out, int out_size, void* d_ws, size_t ws_size,
                              hipStream_t stream) {
  (void)in_sizes; (void)n_in; (void)out_size; (void)ws_size;

  const float* In = (const float*)d_in[0];   // [8,2048,512] f32
  const float* Wq = (const float*)d_in[1];   // [512,8] f32
  const float* Wk = (const float*)d_in[2];   // [512,8] f32
  const float* Wv = (const float*)d_in[3];   // [512,512] f32
  const int*   Pw = (const int*)d_in[4];     // power (=8)
  float* Out = (float*)d_out;                // [8,2048,512] f32

  // Workspace layout (bytes), 256-aligned blocks:
  char* w = (char*)d_ws;
  __bf16* Wvt  = (__bf16*)(w);                 // 512*512*2   = 512 KB
  float*  Qws  = (float*)(w + 524288);         // 16384*8*4   = 512 KB
  float*  Kws  = (float*)(w + 1048576);        // 16384*8*4   = 512 KB
  float*  Mrow = (float*)(w + 1572864);        // 16384*4     =  64 KB
  float*  Li   = (float*)(w + 1638400);        // 16384*4     =  64 KB
  __bf16* Vt   = (__bf16*)(w + 1703936);       // 8*512*2048*2 = 16 MB

  prep_wvt<<<1024, 256, 0, stream>>>(Wv, Wvt);
  qk_kernel<<<512, 256, 0, stream>>>(In, Wq, Wk, Qws, Kws);
  gemm_v<<<dim3(128, 8), 256, 0, stream>>>(In, Wvt, Vt);
  stats_kernel<<<2048, 256, 0, stream>>>(Qws, Kws, Pw, Mrow, Li);
  pv_kernel<<<1024, 256, 0, stream>>>(Qws, Kws, Vt, Mrow, Li, Pw, Out);
}
